// RNN_9010841387087
// MI455X (gfx1250) — matrix operand order repaired and checked by reading, not compiled, not run on hardware
//
#include <hip/hip_runtime.h>

// ---------------------------------------------------------------------------
// Fused 2-layer LSTM for MI455X (gfx1250, wave32, WMMA).
//   B=4096, T=512, D=28, H=32, gates=4H=128.
// One wave owns a 16-row batch tile; both layers advance in lockstep over t,
// so the [B,T,H] intermediate (0.5 GB of traffic) is never materialized.
// Per step per layer: 8x v_wmma_f32_16x16x32_f16 for the input term chained
// into 8x more for the recurrent term (f32 accumulate), then gate math in
// accumulator layout, h re-swizzled C-layout -> A-layout through 1KB of LDS.
// Weights are DMA'd once per block with the Tensor Data Mover.
// Activations use the CDNA5 hardware v_tanh_f32 (sigmoid via tanh identity)
// to minimize the serial critical path of the recurrence (latency-bound at
// ~1-2 waves/SIMD occupancy).
// ---------------------------------------------------------------------------

#define Bsz  4096
#define Tlen 512
#define Din  28
#define Hdim 32
#define WSTRIDE 40   // padded f16 elements per gate-row in LDS (16B aligned, bank-friendly)

typedef __attribute__((ext_vector_type(16))) _Float16 v16h;
typedef __attribute__((ext_vector_type(8)))  _Float16 v8h;
typedef __attribute__((ext_vector_type(8)))  float    v8f;
typedef __attribute__((ext_vector_type(4)))  unsigned v4u;
typedef __attribute__((ext_vector_type(8)))  int      v8i;
typedef __attribute__((ext_vector_type(4)))  int      v4i;

// --- Fast activations: hardware v_tanh_f32 when available, branchless exp form otherwise.
#if __has_builtin(__builtin_amdgcn_tanhf)
__device__ __forceinline__ float tanh_hw(float x) { return __builtin_amdgcn_tanhf(x); }
#else
__device__ __forceinline__ float tanh_hw(float x) {
  // branchless: tanh(x) = 1 - 2/(exp(2x)+1); exp handles +-inf saturation
  float e = __expf(2.0f * x);
  return 1.0f - 2.0f / (e + 1.0f);
}
#endif
__device__ __forceinline__ float sigf(float x) { return 0.5f + 0.5f * tanh_hw(0.5f * x); }

// --- TDM: 2D tensor (dim1 rows x dim0 contiguous f32) -> LDS, one descriptor.
__device__ __forceinline__ void tdm_load_2d(void* lds_dst, const void* gsrc,
                                            unsigned dim0, unsigned dim1) {
  unsigned lds_off = (unsigned)(unsigned long long)lds_dst;  // low 32 bits = LDS byte offset
  unsigned long long ga = (unsigned long long)gsrc;
  v4u g0;
  g0[0] = 1u;                                               // count=1, user descriptor
  g0[1] = lds_off;                                          // lds_addr
  g0[2] = (unsigned)ga;                                     // global_addr[31:0]
  g0[3] = (unsigned)((ga >> 32) & 0x1FFFFFFull) | (2u << 30); // addr[56:32] | type=2
  v8i g1;
  g1[0] = (int)(2u << 16);                                  // data_size = 4B
  g1[1] = (int)((dim0 & 0xFFFFu) << 16);                    // tensor_dim0[15:0]
  g1[2] = (int)((dim0 >> 16) | ((dim1 & 0xFFFFu) << 16));   // dim0 hi | tensor_dim1 lo
  g1[3] = (int)((dim1 >> 16) | ((dim0 & 0xFFFFu) << 16));   // dim1 hi | tile_dim0
  g1[4] = (int)(dim1 & 0xFFFFu);                            // tile_dim1 (tile_dim2 = 0)
  g1[5] = (int)dim0;                                        // tensor_dim0_stride (contiguous)
  g1[6] = 0; g1[7] = 0;
  v4i z4 = {0, 0, 0, 0};                                    // 2D tensor: groups 2/3 unused
  v8i z8 = {0, 0, 0, 0, 0, 0, 0, 0};
  __builtin_amdgcn_tensor_load_to_lds(g0, g1, z4, z4, z8, 0);
}

// --- B-fragment (32x16, f16) from LDS weight array with padded row stride.
// B layout: lanes 0-15 kh=0, lanes 16-31 kh=8; elems 0-7: K=kh+e, 8-15: K=16+kh+e; N = ntile*16 + (lane&15).
__device__ __forceinline__ v16h ld_bfrag(const _Float16* w, int nt, int lane) {
  int n  = nt * 16 + (lane & 15);
  int kb = (lane >> 4) * 8;
  const _Float16* p = w + n * WSTRIDE + kb;
  v8h lo = *(const v8h*)(p);
  v8h hi = *(const v8h*)(p + 16);
  v16h r;
#pragma unroll
  for (int e = 0; e < 8; ++e) { r[e] = lo[e]; r[e + 8] = hi[e]; }
  return r;
}

// --- A-fragment (16x32, f16) from a row-major [16][32] f16 LDS stage.
// A layout: lanes 0-15 kh=0, lanes 16-31 kh=8; elems 0-7: K=kh+e, 8-15: K=16+kh+e.
__device__ __forceinline__ v16h ld_afrag(const _Float16* hs, int m, int kh) {
  v8h lo = *(const v8h*)(hs + m * Hdim + kh);
  v8h hi = *(const v8h*)(hs + m * Hdim + 16 + kh);
  v16h r;
#pragma unroll
  for (int e = 0; e < 8; ++e) { r[e] = lo[e]; r[e + 8] = hi[e]; }
  return r;
}

// --- One LSTM step for one 16-row tile: gates = inA*Wih + hA*Whh (f32 accum),
// activations in accumulator layout, c updated in-register, new h staged
// through LDS to rebuild the A-fragment.
__device__ __forceinline__ void lstm_step(const v16h& inA, v16h& hA, v8f c[2],
                                          const _Float16* wih, const _Float16* whh,
                                          _Float16* hs, int lane) {
  int m = lane & 15, kh = (lane >> 4) * 8;
  v8f d[8];
#pragma unroll
  for (int nt = 0; nt < 8; ++nt) {
    v16h bi = ld_bfrag(wih, nt, lane);
    v16h bh = ld_bfrag(whh, nt, lane);
    v8f z = {0.f, 0.f, 0.f, 0.f, 0.f, 0.f, 0.f, 0.f};
    v8f a0 = __builtin_amdgcn_wmma_f32_16x16x32_f16(false, inA, false, bi, (short)0, z, false, false);
    d[nt]  = __builtin_amdgcn_wmma_f32_16x16x32_f16(false, hA,  false, bh, (short)0, a0, false, false);
  }
  // gates: N-tiles 0,1 = i; 2,3 = f; 4,5 = g; 6,7 = o.  C/D elem (v,lane):
  // row = v + (lane<16 ? 0 : 8), col = ntile*16 + (lane&15)  -> matches c tiles.
#pragma unroll
  for (int n = 0; n < 2; ++n) {
    v8f cc = c[n];
#pragma unroll
    for (int e = 0; e < 8; ++e) {
      float iv = sigf(d[n][e]);
      float fv = sigf(d[n + 2][e]);
      float gv = tanh_hw(d[n + 4][e]);
      float ov = sigf(d[n + 6][e]);
      float cv = fv * cc[e] + iv * gv;
      cc[e] = cv;
      float hv = ov * tanh_hw(cv);
      int row = (lane < 16) ? e : (e + 8);
      hs[row * Hdim + n * 16 + (lane & 15)] = (_Float16)hv;
    }
    c[n] = cc;
  }
  // intra-wave LDS RAW: stores above feed cross-lane loads below
  asm volatile("s_wait_dscnt 0" ::: "memory");
  hA = ld_afrag(hs, m, kh);
}

__global__ __launch_bounds__(256) void lstm_fused(
    const float* __restrict__ x,
    const float* __restrict__ w_ih0, const float* __restrict__ w_hh0,
    const float* __restrict__ w_ih1, const float* __restrict__ w_hh1,
    const float* __restrict__ w_out, const float* __restrict__ b_out,
    float* __restrict__ out) {
  __shared__ float    wstage[128 * Din + 3 * 128 * Hdim];   // f32 TDM landing pad (~62KB)
  __shared__ _Float16 wf16[4][128 * WSTRIDE];               // f16 weights, zero-padded K
  __shared__ _Float16 hst[8][16 * Hdim];                    // per-wave h re-swizzle stage

  // ---- Tensor Data Mover: DMA all four weight matrices into LDS (wave 0 only;
  // TDM ignores EXEC, other waves' scalar branch skips issue entirely).
  if (threadIdx.x == 0) {
    tdm_load_2d(&wstage[0],                           w_ih0, Din,  128);
    tdm_load_2d(&wstage[128 * Din],                   w_hh0, Hdim, 128);
    tdm_load_2d(&wstage[128 * Din + 128 * Hdim],      w_ih1, Hdim, 128);
    tdm_load_2d(&wstage[128 * Din + 2 * 128 * Hdim],  w_hh1, Hdim, 128);
    __builtin_amdgcn_s_wait_tensorcnt(0);
  }
  __syncthreads();

  // ---- Convert f32 -> f16 with K padded to WSTRIDE (zeros beyond D/H).
  for (int task = threadIdx.x; task < 512; task += 256) {
    int mat = task >> 7, r = task & 127;
    int C    = (mat == 0) ? Din : Hdim;
    int base = (mat == 0) ? 0 : (128 * Din + (mat - 1) * 128 * Hdim);
    const float* src = &wstage[base + r * C];
    _Float16* dst = &wf16[mat][r * WSTRIDE];
    for (int k = 0; k < WSTRIDE; ++k) dst[k] = (k < C) ? (_Float16)src[k] : (_Float16)0.f;
  }
  __syncthreads();

  const int lane = threadIdx.x & 31;
  const int wave = threadIdx.x >> 5;
  const int m  = lane & 15;
  const int kh = (lane >> 4) * 8;
  const int b0 = blockIdx.x * 128 + wave * 16;
  const float* xrow = x + (size_t)(b0 + m) * (Tlen * Din);
  _Float16* hs = hst[wave];

  v16h h1A, h2A;
#pragma unroll
  for (int e = 0; e < 16; ++e) { h1A[e] = (_Float16)0.f; h2A[e] = (_Float16)0.f; }
  v8f c1[2], c2[2];
#pragma unroll
  for (int n = 0; n < 2; ++n) {
#pragma unroll
    for (int e = 0; e < 8; ++e) { c1[n][e] = 0.f; c2[n][e] = 0.f; }
  }

  for (int t = 0; t < Tlen; ++t) {
    const float* xp = xrow + t * Din;
    if (t + 8 < Tlen) __builtin_prefetch(xp + 8 * Din, 0, 1);   // global_prefetch ahead
    // Assemble x A-fragment: elems 0-7 = x[m][kh+e]; elems 8-15 = x[m][16+kh+e]
    // (K=28..31 zero-padded; all float4 loads 16B-aligned, in-bounds).
    float4 fa = *(const float4*)(xp + kh);
    float4 fb = *(const float4*)(xp + kh + 4);
    float4 fc = *(const float4*)(xp + 16 + kh);
    float4 fd = make_float4(0.f, 0.f, 0.f, 0.f);
    if (kh == 0) fd = *(const float4*)(xp + 20);                // kh==8 -> K=28..31 are pad
    v16h xA;
    xA[0]  = (_Float16)fa.x; xA[1]  = (_Float16)fa.y; xA[2]  = (_Float16)fa.z; xA[3]  = (_Float16)fa.w;
    xA[4]  = (_Float16)fb.x; xA[5]  = (_Float16)fb.y; xA[6]  = (_Float16)fb.z; xA[7]  = (_Float16)fb.w;
    xA[8]  = (_Float16)fc.x; xA[9]  = (_Float16)fc.y; xA[10] = (_Float16)fc.z; xA[11] = (_Float16)fc.w;
    xA[12] = (_Float16)fd.x; xA[13] = (_Float16)fd.y; xA[14] = (_Float16)fd.z; xA[15] = (_Float16)fd.w;

    lstm_step(xA,  h1A, c1, wf16[0], wf16[1], hs, lane);   // layer 1
    lstm_step(h1A, h2A, c2, wf16[2], wf16[3], hs, lane);   // layer 2 (fused, lockstep)
  }

  // ---- Output projection: hs holds the final h2 tile (f16, row-major [16][32]).
  if (lane < 16) {
    int bm = b0 + lane;
#pragma unroll
    for (int j = 0; j < 10; ++j) {
      float acc = b_out[j];
#pragma unroll
      for (int k = 0; k < Hdim; ++k)
        acc += (float)hs[lane * Hdim + k] * w_out[j * Hdim + k];
      out[bm * 10 + j] = acc;
    }
  }
}

extern "C" void kernel_launch(void* const* d_in, const int* in_sizes, int n_in,
                              void* d_out, int out_size, void* d_ws, size_t ws_size,
                              hipStream_t stream) {
  (void)in_sizes; (void)n_in; (void)out_size; (void)d_ws; (void)ws_size;
  const float* x     = (const float*)d_in[0];
  const float* w_ih0 = (const float*)d_in[1];
  const float* w_hh0 = (const float*)d_in[2];
  const float* w_ih1 = (const float*)d_in[3];
  const float* w_hh1 = (const float*)d_in[4];
  const float* w_out = (const float*)d_in[5];
  const float* b_out = (const float*)d_in[6];
  dim3 grid(Bsz / 128), block(256);   // 8 waves/block, one 16-row tile per wave
  lstm_fused<<<grid, block, 0, stream>>>(x, w_ih0, w_hh0, w_ih1, w_hh1,
                                         w_out, b_out, (float*)d_out);
}
